// MixtureOfExperts_72808285602477
// MI455X (gfx1250) — matrix operand order repaired and checked by reading, not compiled, not run on hardware
//
#include <hip/hip_runtime.h>
#include <math.h>

#define B_TOK   8192
#define D_IN    1024
#define D_OUT   1024
#define NEXP    16
#define EXP_H   4096
#define GATE_H  512
#define GATE_H2 256
#define WTHRESH 0.01f

#define TM 128
#define TN 128
#define TK 64                 // K-slab per LDS stage (2 WMMA k-steps)
#define CAP   18432           // 144 * 128 row slots (>= worst-case padded total)
#define NTILE (CAP / TM)      // 144

typedef __bf16 bf16;
typedef __bf16 bf16x4  __attribute__((ext_vector_type(4)));
typedef __bf16 bf16x8  __attribute__((ext_vector_type(8)));
typedef __bf16 bf16x16 __attribute__((ext_vector_type(16)));
typedef float  f32x8   __attribute__((ext_vector_type(8)));

// hardware RNE conversion (v_cvt_pk_bf16_f32 on gfx1250)
__device__ inline bf16 to_bf16(float f) { return (bf16)f; }

__device__ inline bf16x16 cat16(const bf16x8 a, const bf16x8 b) {
    return __builtin_shufflevector(a, b, 0,1,2,3,4,5,6,7,8,9,10,11,12,13,14,15);
}

// ---------------------------------------------------------------- zero output
__global__ __launch_bounds__(256) void k_zero_out(float4* out, int n4) {
    int i = blockIdx.x * 256 + threadIdx.x;
    if (i < n4) out[i] = make_float4(0.f, 0.f, 0.f, 0.f);
}

// ---------------------------------------------------------- routing meta init
__global__ __launch_bounds__(256) void k_init_routing(int* counts, int* cursor,
                                                      int* tok, float* wt) {
    int i = blockIdx.x * 256 + threadIdx.x;
    if (i < CAP) { tok[i] = 0; wt[i] = 0.f; }
    if (i < NEXP) { counts[i] = 0; cursor[i] = 0; }
}

// ------------------------------------------------------------------- gating
// 8 tokens per 256-thread block: fp32 VALU (keeps softmax/top-k selection exact)
#define GTOK 8
__global__ __launch_bounds__(256) void k_gating(
    const float* __restrict__ x,
    const float* __restrict__ gW1, const float* __restrict__ gb1,
    const float* __restrict__ gW2, const float* __restrict__ gb2,
    const float* __restrict__ gW3, const float* __restrict__ gb3,
    const float* __restrict__ dW,  const float* __restrict__ db,
    int* __restrict__ topi, float* __restrict__ topw)
{
    __shared__ float sx[GTOK][D_IN];
    __shared__ float sh1[GTOK][GATE_H];
    __shared__ float sh2[GTOK][GATE_H2];
    __shared__ float slog[GTOK][NEXP];

    const int t = threadIdx.x;
    const int tok0 = blockIdx.x * GTOK;

    const float4* xin = (const float4*)(x + (size_t)tok0 * D_IN);
    float4* sxv = (float4*)&sx[0][0];
    for (int i = t; i < GTOK * D_IN / 4; i += 256) sxv[i] = xin[i];
    __syncthreads();

    for (int j = t; j < GATE_H; j += 256) {
        float a[GTOK]; for (int q = 0; q < GTOK; ++q) a[q] = 0.f;
        for (int k = 0; k < D_IN; ++k) {
            float wv = gW1[k * GATE_H + j];
            for (int q = 0; q < GTOK; ++q) a[q] += sx[q][k] * wv;
        }
        float bb = gb1[j];
        for (int q = 0; q < GTOK; ++q) sh1[q][j] = fmaxf(a[q] + bb, 0.f);
    }
    __syncthreads();

    {
        int j = t;  // 256 cols
        float a[GTOK]; for (int q = 0; q < GTOK; ++q) a[q] = 0.f;
        for (int k = 0; k < GATE_H; ++k) {
            float wv = gW2[k * GATE_H2 + j];
            for (int q = 0; q < GTOK; ++q) a[q] += sh1[q][k] * wv;
        }
        float bb = gb2[j];
        for (int q = 0; q < GTOK; ++q) sh2[q][j] = fmaxf(a[q] + bb, 0.f);
    }
    __syncthreads();

    if (t < GTOK * NEXP) {
        int q = t >> 4, e = t & 15;
        float a = 0.f;
        for (int k = 0; k < GATE_H2; ++k) a += sh2[q][k] * gW3[k * NEXP + e];
        float d = 0.f;
        for (int k = 0; k < D_IN; ++k) d += sx[q][k] * dW[k * NEXP + e];
        slog[q][e] = a + gb3[e] + 0.1f * (d + db[e]);
    }
    __syncthreads();

    if (t < GTOK) {
        float m = -1e30f;
        for (int e = 0; e < NEXP; ++e) m = fmaxf(m, slog[t][e]);
        float g[NEXP], s = 0.f;
        for (int e = 0; e < NEXP; ++e) { g[e] = __expf(slog[t][e] - m); s += g[e]; }
        float inv = 1.f / s;
        int i0 = 0, i1 = 1; float v0 = -1.f, v1 = -1.f;
        for (int e = 0; e < NEXP; ++e) {
            float ge = g[e] * inv;
            if (ge > v0)      { v1 = v0; i1 = i0; v0 = ge; i0 = e; }
            else if (ge > v1) { v1 = ge; i1 = e; }
        }
        float e1 = __expf(v1 - v0);
        float w0 = 1.f / (1.f + e1);
        float w1 = e1 / (1.f + e1);
        if (w0 <= WTHRESH) w0 = 0.f;
        if (w1 <= WTHRESH) w1 = 0.f;
        int id = tok0 + t;
        topi[id * 2 + 0] = i0; topw[id * 2 + 0] = w0;
        topi[id * 2 + 1] = i1; topw[id * 2 + 1] = w1;
    }
}

// ------------------------------------------------------------ count / prefix / scatter
__global__ __launch_bounds__(256) void k_count(const int* topi, const float* topw,
                                               int* counts, int n) {
    int i = blockIdx.x * 256 + threadIdx.x;
    if (i >= n) return;
    float w = topw[i];
    if (w > WTHRESH) atomicAdd(&counts[topi[i]], 1);
}

__global__ void k_prefix(const int* counts, int* cursor, int* segoff) {
    if (threadIdx.x == 0) {
        int run = 0;
        for (int e = 0; e < NEXP; ++e) {
            segoff[e] = run;
            cursor[e] = run;
            run += (counts[e] + TM - 1) & ~(TM - 1);   // 128-aligned segments
        }
        segoff[NEXP] = run;
    }
}

__global__ __launch_bounds__(256) void k_scatter(const int* topi, const float* topw,
                                                 int* cursor, int* tok, float* wt, int n) {
    int i = blockIdx.x * 256 + threadIdx.x;
    if (i >= n) return;
    float w = topw[i];
    if (w > WTHRESH) {
        int e = topi[i];
        int slot = atomicAdd(&cursor[e], 1);
        tok[slot] = i >> 1;
        wt[slot] = w;
    }
}

// --------------------------------------------------------------------------
// B-tile staging: W row-major fp32 -> LDS [n][k] bf16.
// Each thread owns an 8n x 4k sub-block: 4 coalesced row reads, then the 4
// k-values per n are contiguous in LDS -> one ds_store_b64 per n.
__device__ inline void stage_B(const float* __restrict__ W, int ldw,
                               int k0, int n_base, bf16* __restrict__ sB, int t)
{
    const int kg = t >> 4;            // 0..15, 4 k each
    const int n  = (t & 15) * 8;      // 0..120
    bf16 vals[4][8];
    #pragma unroll
    for (int kr = 0; kr < 4; ++kr) {
        const float4* src = (const float4*)(W + (size_t)(k0 + kg * 4 + kr) * ldw + n_base + n);
        float4 b0 = src[0], b1 = src[1];
        vals[kr][0] = to_bf16(b0.x); vals[kr][1] = to_bf16(b0.y);
        vals[kr][2] = to_bf16(b0.z); vals[kr][3] = to_bf16(b0.w);
        vals[kr][4] = to_bf16(b1.x); vals[kr][5] = to_bf16(b1.y);
        vals[kr][6] = to_bf16(b1.z); vals[kr][7] = to_bf16(b1.w);
    }
    #pragma unroll
    for (int i = 0; i < 8; ++i) {
        bf16x4 p;
        p[0] = vals[0][i]; p[1] = vals[1][i]; p[2] = vals[2][i]; p[3] = vals[3][i];
        *(bf16x4*)&sB[(n + i) * TK + kg * 4] = p;
    }
}

// ------------------------------------------------------------- FC1 (x -> relu -> H)
// grid: (NTILE, EXP_H/TN). 256 threads = 8 waves; wave (wr,wc) owns 64x32.
__global__ __launch_bounds__(256) void k_fc1(
    const float* __restrict__ x, const float* __restrict__ eW1,
    const float* __restrict__ eb1, bf16* __restrict__ H,
    const int* __restrict__ tok, const int* __restrict__ segoff)
{
    const int row_base = blockIdx.x * TM;
    if (row_base >= segoff[NEXP]) return;
    int e = 0;
    for (int i = 1; i < NEXP; ++i) if (row_base >= segoff[i]) e = i;

    const int n_base = blockIdx.y * TN;
    const float* W = eW1 + (size_t)e * D_IN * EXP_H;   // [1024][4096]
    const float* bias = eb1 + (size_t)e * EXP_H;

    __shared__ bf16 sA[TM * TK];   // [m][k]
    __shared__ bf16 sB[TN * TK];   // [n][k]  (B^T tile)
    __shared__ int  sTok[TM];

    const int t = threadIdx.x;
    if (t < TM) sTok[t] = tok[row_base + t];
    __syncthreads();

    const int lane = t & 31, wid = t >> 5;
    const int wr = wid >> 2, wc = wid & 3;
    const int lh = lane >> 4;        // lane half
    const int ll = lane & 15;

    f32x8 acc[4][2];
    for (int mi = 0; mi < 4; ++mi)
        for (int ni = 0; ni < 2; ++ni)
            for (int r = 0; r < 8; ++r) acc[mi][ni][r] = 0.f;

    for (int k0 = 0; k0 < D_IN; k0 += TK) {
        // prefetch next W k-slab (global_prefetch_b8)
        if (k0 + TK < D_IN)
            __builtin_prefetch(W + (size_t)(k0 + TK) * EXP_H + n_base + (t & 63) * 2, 0, 1);

        // A: gather token rows, fp32 -> bf16 (hardware cvt), vector LDS store
        for (int j = 0; j < 4; ++j) {
            int c = t * 4 + j;                  // 1024 chunks of 8
            int row = c >> 3, kk = (c & 7) * 8;
            const float4* src = (const float4*)(x + (size_t)sTok[row] * D_IN + k0 + kk);
            float4 a0 = src[0], a1 = src[1];
            bf16x8 v;
            v[0] = to_bf16(a0.x); v[1] = to_bf16(a0.y); v[2] = to_bf16(a0.z); v[3] = to_bf16(a0.w);
            v[4] = to_bf16(a1.x); v[5] = to_bf16(a1.y); v[6] = to_bf16(a1.z); v[7] = to_bf16(a1.w);
            *(bf16x8*)&sA[row * TK + kk] = v;
        }
        // B: packed transpose staging (ds_store_b64)
        stage_B(W, EXP_H, k0, n_base, sB, t);
        __syncthreads();

        #pragma unroll
        for (int ks = 0; ks < 2; ++ks) {
            const int kb = ks * 32;
            bf16x16 af[4];
            for (int mi = 0; mi < 4; ++mi) {
                int m = wr * 64 + mi * 16 + ll;
                af[mi] = cat16(*(const bf16x8*)&sA[m * TK + kb + lh * 8],
                               *(const bf16x8*)&sA[m * TK + kb + 16 + lh * 8]);
            }
            bf16x16 bfm[2];
            for (int ni = 0; ni < 2; ++ni) {
                int n = wc * 32 + ni * 16 + ll;
                bfm[ni] = cat16(*(const bf16x8*)&sB[n * TK + kb + lh * 8],
                                *(const bf16x8*)&sB[n * TK + kb + 16 + lh * 8]);
            }
            for (int mi = 0; mi < 4; ++mi)
                for (int ni = 0; ni < 2; ++ni)
                    acc[mi][ni] = __builtin_amdgcn_wmma_f32_16x16x32_bf16(
                        false, af[mi], false, bfm[ni], (short)0, acc[mi][ni], false, false);
        }
        __syncthreads();
    }

    for (int mi = 0; mi < 4; ++mi)
        for (int ni = 0; ni < 2; ++ni) {
            int n = n_base + wc * 32 + ni * 16 + ll;
            float bb = bias[n];
            for (int r = 0; r < 8; ++r) {
                int m = wr * 64 + mi * 16 + r + lh * 8;
                float v = acc[mi][ni][r] + bb;
                H[(size_t)(row_base + m) * EXP_H + n] = to_bf16(v > 0.f ? v : 0.f);
            }
        }
}

// ------------------------------------------------------------- FC2 (H -> out, fused combine)
// grid: (NTILE, D_OUT/TN)
__global__ __launch_bounds__(256) void k_fc2(
    const bf16* __restrict__ H, const float* __restrict__ eW2,
    const float* __restrict__ eb2, const int* __restrict__ tok,
    const float* __restrict__ wt, const int* __restrict__ segoff,
    float* __restrict__ out)
{
    const int row_base = blockIdx.x * TM;
    if (row_base >= segoff[NEXP]) return;
    int e = 0;
    for (int i = 1; i < NEXP; ++i) if (row_base >= segoff[i]) e = i;

    const int n_base = blockIdx.y * TN;
    const float* W = eW2 + (size_t)e * EXP_H * D_OUT;  // [4096][1024]
    const float* bias = eb2 + (size_t)e * D_OUT;

    __shared__ bf16  sA[TM * TK];
    __shared__ bf16  sB[TN * TK];
    __shared__ int   sTok[TM];
    __shared__ float sW[TM];

    const int t = threadIdx.x;
    if (t < TM) { sTok[t] = tok[row_base + t]; sW[t] = wt[row_base + t]; }
    __syncthreads();

    const int lane = t & 31, wid = t >> 5;
    const int wr = wid >> 2, wc = wid & 3;
    const int lh = lane >> 4, ll = lane & 15;

    f32x8 acc[4][2];
    for (int mi = 0; mi < 4; ++mi)
        for (int ni = 0; ni < 2; ++ni)
            for (int r = 0; r < 8; ++r) acc[mi][ni][r] = 0.f;

    for (int k0 = 0; k0 < EXP_H; k0 += TK) {
        if (k0 + TK < EXP_H)
            __builtin_prefetch(W + (size_t)(k0 + TK) * D_OUT + n_base + (t & 63) * 2, 0, 1);

        // A: bf16 hidden rows (contiguous slots) — pure b128 copies
        for (int j = 0; j < 4; ++j) {
            int c = t * 4 + j;
            int row = c >> 3, kk = (c & 7) * 8;
            *(bf16x8*)&sA[row * TK + kk] =
                *(const bf16x8*)(H + (size_t)(row_base + row) * EXP_H + k0 + kk);
        }
        // B: packed transpose staging (ds_store_b64)
        stage_B(W, D_OUT, k0, n_base, sB, t);
        __syncthreads();

        #pragma unroll
        for (int ks = 0; ks < 2; ++ks) {
            const int kb = ks * 32;
            bf16x16 af[4];
            for (int mi = 0; mi < 4; ++mi) {
                int m = wr * 64 + mi * 16 + ll;
                af[mi] = cat16(*(const bf16x8*)&sA[m * TK + kb + lh * 8],
                               *(const bf16x8*)&sA[m * TK + kb + 16 + lh * 8]);
            }
            bf16x16 bfm[2];
            for (int ni = 0; ni < 2; ++ni) {
                int n = wc * 32 + ni * 16 + ll;
                bfm[ni] = cat16(*(const bf16x8*)&sB[n * TK + kb + lh * 8],
                                *(const bf16x8*)&sB[n * TK + kb + 16 + lh * 8]);
            }
            for (int mi = 0; mi < 4; ++mi)
                for (int ni = 0; ni < 2; ++ni)
                    acc[mi][ni] = __builtin_amdgcn_wmma_f32_16x16x32_bf16(
                        false, af[mi], false, bfm[ni], (short)0, acc[mi][ni], false, false);
        }
        __syncthreads();
    }

    for (int mi = 0; mi < 4; ++mi)
        for (int ni = 0; ni < 2; ++ni) {
            int n = n_base + wc * 32 + ni * 16 + ll;
            float bb = bias[n];
            for (int r = 0; r < 8; ++r) {
                int m = wr * 64 + mi * 16 + r + lh * 8;
                float wgt = sW[m];
                atomicAdd(out + (size_t)sTok[m] * D_OUT + n, wgt * (acc[mi][ni][r] + bb));
            }
        }
}

// ---------------------------------------------------------------------- launch
extern "C" void kernel_launch(void* const* d_in, const int* in_sizes, int n_in,
                              void* d_out, int out_size, void* d_ws, size_t ws_size,
                              hipStream_t stream) {
    const float* x   = (const float*)d_in[0];
    const float* gW1 = (const float*)d_in[1];
    const float* gb1 = (const float*)d_in[2];
    const float* gW2 = (const float*)d_in[3];
    const float* gb2 = (const float*)d_in[4];
    const float* gW3 = (const float*)d_in[5];
    const float* gb3 = (const float*)d_in[6];
    const float* dW  = (const float*)d_in[7];
    const float* db  = (const float*)d_in[8];
    const float* eW1 = (const float*)d_in[9];
    const float* eb1 = (const float*)d_in[10];
    const float* eW2 = (const float*)d_in[11];
    const float* eb2 = (const float*)d_in[12];
    float* out = (float*)d_out;

    // workspace layout
    char* ws = (char*)d_ws;
    bf16*  Hbuf   = (bf16*)ws;                                  // CAP*4096 bf16
    size_t o      = (size_t)CAP * EXP_H * sizeof(bf16);
    int*   counts = (int*)(ws + o);      o += 64;
    int*   cursor = (int*)(ws + o);      o += 64;
    int*   segoff = (int*)(ws + o);      o += 128;
    int*   tok    = (int*)(ws + o);      o += (size_t)CAP * 4;
    float* wt     = (float*)(ws + o);    o += (size_t)CAP * 4;
    int*   topi   = (int*)(ws + o);      o += (size_t)B_TOK * 2 * 4;
    float* topw   = (float*)(ws + o);

    const int n4 = B_TOK * D_OUT / 4;
    k_zero_out<<<(n4 + 255) / 256, 256, 0, stream>>>((float4*)out, n4);
    k_init_routing<<<(CAP + 255) / 256, 256, 0, stream>>>(counts, cursor, tok, wt);

    k_gating<<<B_TOK / GTOK, 256, 0, stream>>>(x, gW1, gb1, gW2, gb2, gW3, gb3,
                                               dW, db, topi, topw);

    const int npair = B_TOK * 2;
    k_count<<<(npair + 255) / 256, 256, 0, stream>>>(topi, topw, counts, npair);
    k_prefix<<<1, 32, 0, stream>>>(counts, cursor, segoff);
    k_scatter<<<(npair + 255) / 256, 256, 0, stream>>>(topi, topw, cursor, tok, wt, npair);

    dim3 g1(NTILE, EXP_H / TN);   // (144, 32)
    k_fc1<<<g1, 256, 0, stream>>>(x, eW1, eb1, Hbuf, tok, segoff);

    dim3 g2(NTILE, D_OUT / TN);   // (144, 8)
    k_fc2<<<g2, 256, 0, stream>>>(Hbuf, eW2, eb2, tok, wt, segoff, out);
}
